// GraphTransformerDecode_67568425501198
// MI455X (gfx1250) — matrix operand order repaired and checked by reading, not compiled, not run on hardware
//
#include <hip/hip_runtime.h>

typedef __bf16 bf16_t;
typedef __bf16 bv16 __attribute__((ext_vector_type(16)));
typedef float  v8f  __attribute__((ext_vector_type(8)));

#define DD 256
#define NB 4
#define SLR 12         // 3*NB valid SL rows
#define O2R 100        // 25*NB valid O2 rows
#define O2P 112        // padded to 7 M-tiles
#define SS 516         // fp32 stage row stride (wide layouts, 16 rows)
#define SS4 260        // fp32 stage row stride (G4 chunks, 32 rows)

// d_ws bf16 layout (element offsets): 7 transposed 256x256 matrices, in order:
// Wq1^T, Wv1^T, Wo1^T, Wq2^T, Wk2^T, Wv2^T, Wo2^T
#define OFF_WT1   0        // [512][256] = [Wq1|Wv1]^T
#define OFF_WOT1  131072   // [256][256] = Wo1^T
#define OFF_WQ2T  196608   // [256][256] = Wq2^T
#define OFF_WKV2T 262144   // [512][256] = [Wk2|Wv2]^T
#define OFF_WO2T  393216   // [256][256] = Wo2^T
#define WS_ELEMS  458752

#define SMEM_BYTES 153344

__device__ __forceinline__ v8f wmma_bf16(bv16 a, bv16 b, v8f c) {
    return __builtin_amdgcn_wmma_f32_16x16x32_bf16(false, a, false, b, (short)0, c, false, false);
}

// A fragment (16x32 bf16) from row-major [rows][256] buffer.
// ISA layout: lanes 0-15: M=lane, K = k0+{0..7, 16..23}; lanes 16-31: M=lane-16, K = k0+{8..15, 24..31}
__device__ __forceinline__ bv16 ld_a(const bf16_t* A, int m0, int k0, int lane) {
    int m  = m0 + (lane & 15);
    int kk = k0 + ((lane >> 4) << 3);
    const bf16_t* p = A + m * DD + kk;
    union { bv16 v; uint4 q[2]; } r;
    r.q[0] = *(const uint4*)(p);
    r.q[1] = *(const uint4*)(p + 16);
    return r.v;
}

// B fragment (32x16 bf16) from transposed weights W^T [N][256]; Wt pre-offset to the n-tile.
// Layout: lane L holds column n = L&15, K = k0 + (L>=16 ? 8 : 0) + {0..7, 16..23}
__device__ __forceinline__ bv16 ld_b(const bf16_t* Wt, int k0, int lane) {
    int n  = lane & 15;
    int kk = k0 + ((lane >> 4) << 3);
    const bf16_t* p = Wt + n * DD + kk;
    union { bv16 v; uint4 q[2]; } r;
    r.q[0] = *(const uint4*)(p);
    r.q[1] = *(const uint4*)(p + 16);
    return r.v;
}

// One global_prefetch line per lane covers a 16x256 bf16 n-tile (8 KB @ 256B spacing)
__device__ __forceinline__ void pf_tile(const bf16_t* Wt, int lane) {
    __builtin_prefetch((const void*)(Wt + lane * 128), 0, 1);
}

__device__ __forceinline__ void wave_reduce2(float& s, float& q) {
#pragma unroll
    for (int off = 16; off > 0; off >>= 1) {
        s += __shfl_xor(s, off, 32);
        q += __shfl_xor(q, off, 32);
    }
}

// Coalesced 32x32 LDS-tile transpose-convert: ws[m][r][k] = (bf16) Wsrc_m[k][r]
__global__ void __launch_bounds__(256) prep_weights(
    const float* __restrict__ W0, const float* __restrict__ W1, const float* __restrict__ W2,
    const float* __restrict__ W3, const float* __restrict__ W4, const float* __restrict__ W5,
    const float* __restrict__ W6, bf16_t* __restrict__ ws) {
    __shared__ float tile[32][33];
    int m = blockIdx.x >> 6;          // matrix id 0..6
    int t = blockIdx.x & 63;          // 8x8 tiles of 32
    int tx = (t & 7) * 32;            // src col base
    int ty = (t >> 3) * 32;           // src row base
    const float* W = (m == 0) ? W0 : (m == 1) ? W1 : (m == 2) ? W2 :
                     (m == 3) ? W3 : (m == 4) ? W4 : (m == 5) ? W5 : W6;
    int lx = threadIdx.x & 31;
    int ly = threadIdx.x >> 5;
#pragma unroll
    for (int i = 0; i < 32; i += 8)
        tile[ly + i][lx] = W[(size_t)(ty + ly + i) * 256 + tx + lx];
    __syncthreads();
    bf16_t* dst = ws + m * 65536;
#pragma unroll
    for (int i = 0; i < 32; i += 8)
        dst[(size_t)(tx + ly + i) * 256 + ty + lx] = (bf16_t)tile[lx][ly + i];
}

__global__ void __launch_bounds__(256) fused_gtd_kernel(
    const float* __restrict__ X,  const float* __restrict__ S1, const float* __restrict__ S2,
    const float* __restrict__ adj2,
    const float* __restrict__ g11, const float* __restrict__ b11,
    const float* __restrict__ g12, const float* __restrict__ b12,
    const float* __restrict__ g21, const float* __restrict__ b21,
    const float* __restrict__ g22, const float* __restrict__ b22,
    const bf16_t* __restrict__ wts, float* __restrict__ out) {

    extern __shared__ __align__(16) char smem[];
    float*  stage = (float*)(smem);              // 8320 f (16x516 or 32x260)
    float*  att   = (float*)(smem + 33280);      // [4][4][25][3]
    float*  Gm    = (float*)(smem + 38080);      // [4][4][3][3]
    bf16_t* xbf   = (bf16_t*)(smem + 38656);     // [16][256]
    bf16_t* O1bf  = (bf16_t*)(smem + 46848);     // [16][256]
    bf16_t* SLbf  = (bf16_t*)(smem + 55040);     // [16][256]
    bf16_t* Kgbf  = (bf16_t*)(smem + 63232);     // [16][256]
    bf16_t* SLqbf = (bf16_t*)(smem + 71424);     // [16][256]
    bf16_t* KVpbf = (bf16_t*)(smem + 79616);     // [16][512]  (Kp | Vp)
    bf16_t* O2bf  = (bf16_t*)(smem + 96000);     // [112][256]

    const int tid  = threadIdx.x;
    const int wave = tid >> 5;
    const int lane = tid & 31;
    const int nlo  = lane & 15;
    const int mh   = (lane >> 4) * 8;
    const int b0   = blockIdx.x * NB;

    // ---- S0: load X (bf16), zero-pad rows 4..15 ----
    for (int idx = tid; idx < 16 * DD; idx += 256) {
        int r = idx >> 8, c = idx & 255;
        xbf[idx] = (r < NB) ? (bf16_t)X[(size_t)(b0 + r) * DD + c] : (bf16_t)0.f;
    }
    __syncthreads();

    // ---- S1 (G1): Y[16][512] = xbf @ [Wq1|Wv1]  (32 n-tiles, 4/wave) ----
    {
        const bf16_t* Wt0 = wts + OFF_WT1 + (wave * 4) * 16 * DD;
        v8f acc[4];
#pragma unroll
        for (int t = 0; t < 4; ++t) { acc[t] = {}; pf_tile(Wt0 + t * 16 * DD, lane); }
#pragma unroll
        for (int kk = 0; kk < 8; ++kk) {
            bv16 a = ld_a(xbf, 0, kk * 32, lane);
#pragma unroll
            for (int t = 0; t < 4; ++t) {
                bv16 b = ld_b(Wt0 + t * 16 * DD, kk * 32, lane);
                acc[t] = wmma_bf16(a, b, acc[t]);
            }
        }
#pragma unroll
        for (int t = 0; t < 4; ++t) {
            int n = (wave * 4 + t) * 16 + nlo;
#pragma unroll
            for (int i = 0; i < 8; ++i) stage[(mh + i) * SS + n] = acc[t][i];
        }
    }
    __syncthreads();

    // ---- S2: O1 = S1[i]*yq + v (softmax over 1 key == 1), LN(g11,b11) -> O1bf ----
    for (int r = wave; r < SLR; r += 8) {
        int lb = r / 3, i3 = r - lb * 3;
        float s1 = S1[i3];
        int cb = lane * 8;
        float vals[8]; float s = 0.f, q = 0.f;
#pragma unroll
        for (int j = 0; j < 8; ++j) {
            int c = cb + j;
            float o = s1 * stage[lb * SS + c] + stage[lb * SS + 256 + c];
            vals[j] = o; s += o; q += o * o;
        }
        wave_reduce2(s, q);
        float mean = s * (1.f / 256.f);
        float rinv = rsqrtf(q * (1.f / 256.f) - mean * mean + 1e-5f);
#pragma unroll
        for (int j = 0; j < 8; ++j) {
            int c = cb + j;
            O1bf[r * DD + c] = (bf16_t)((vals[j] - mean) * rinv * g11[c] + b11[c]);
        }
    }
    for (int idx = tid; idx < 4 * DD; idx += 256) O1bf[SLR * DD + idx] = (bf16_t)0.f;
    __syncthreads();

    // ---- S3 (G2): T = O1bf @ Wo1  (16 n-tiles, 2/wave) ----
    {
        const bf16_t* Wt0 = wts + OFF_WOT1 + (wave * 2) * 16 * DD;
        v8f acc[2];
#pragma unroll
        for (int t = 0; t < 2; ++t) { acc[t] = {}; pf_tile(Wt0 + t * 16 * DD, lane); }
#pragma unroll
        for (int kk = 0; kk < 8; ++kk) {
            bv16 a = ld_a(O1bf, 0, kk * 32, lane);
#pragma unroll
            for (int t = 0; t < 2; ++t) {
                bv16 b = ld_b(Wt0 + t * 16 * DD, kk * 32, lane);
                acc[t] = wmma_bf16(a, b, acc[t]);
            }
        }
#pragma unroll
        for (int t = 0; t < 2; ++t) {
            int n = (wave * 2 + t) * 16 + nlo;
#pragma unroll
            for (int i = 0; i < 8; ++i) stage[(mh + i) * SS + n] = acc[t][i];
        }
    }
    __syncthreads();

    // ---- S4: SL = LN(O1 + relu(T), g12,b12) -> SLbf ----
    for (int r = wave; r < SLR; r += 8) {
        int cb = lane * 8;
        float vals[8]; float s = 0.f, q = 0.f;
#pragma unroll
        for (int j = 0; j < 8; ++j) {
            int c = cb + j;
            float o = (float)O1bf[r * DD + c] + fmaxf(stage[r * SS + c], 0.f);
            vals[j] = o; s += o; q += o * o;
        }
        wave_reduce2(s, q);
        float mean = s * (1.f / 256.f);
        float rinv = rsqrtf(q * (1.f / 256.f) - mean * mean + 1e-5f);
#pragma unroll
        for (int j = 0; j < 8; ++j) {
            int c = cb + j;
            SLbf[r * DD + c] = (bf16_t)((vals[j] - mean) * rinv * g12[c] + b12[c]);
        }
    }
    for (int idx = tid; idx < 4 * DD; idx += 256) SLbf[SLR * DD + idx] = (bf16_t)0.f;
    __syncthreads();

    // ---- S5: graph conv Kg = adj2 @ SL ----
    for (int idx = tid; idx < SLR * DD; idx += 256) {
        int r = idx >> 8, c = idx & 255;
        int lb = r / 3, i3 = r - lb * 3;
        const float* a2 = adj2 + i3 * 3;
        float acc = a2[0] * (float)SLbf[(lb * 3 + 0) * DD + c]
                  + a2[1] * (float)SLbf[(lb * 3 + 1) * DD + c]
                  + a2[2] * (float)SLbf[(lb * 3 + 2) * DD + c];
        Kgbf[idx] = (bf16_t)acc;
    }
    for (int idx = tid; idx < 4 * DD; idx += 256) Kgbf[SLR * DD + idx] = (bf16_t)0.f;
    __syncthreads();

    // ---- S6 (G3a): SLq = SLbf @ Wq2 ; (G3b): [Kp|Vp] = Kgbf @ [Wk2|Wv2] ----
    {
        const bf16_t* Wt0 = wts + OFF_WQ2T + (wave * 2) * 16 * DD;
        v8f acc[2];
#pragma unroll
        for (int t = 0; t < 2; ++t) { acc[t] = {}; pf_tile(Wt0 + t * 16 * DD, lane); }
#pragma unroll
        for (int kk = 0; kk < 8; ++kk) {
            bv16 a = ld_a(SLbf, 0, kk * 32, lane);
#pragma unroll
            for (int t = 0; t < 2; ++t) {
                bv16 b = ld_b(Wt0 + t * 16 * DD, kk * 32, lane);
                acc[t] = wmma_bf16(a, b, acc[t]);
            }
        }
#pragma unroll
        for (int t = 0; t < 2; ++t) {
            int n = (wave * 2 + t) * 16 + nlo;
#pragma unroll
            for (int i = 0; i < 8; ++i) SLqbf[(mh + i) * DD + n] = (bf16_t)acc[t][i];
        }
    }
    {
        const bf16_t* Wt0 = wts + OFF_WKV2T + (wave * 4) * 16 * DD;
        v8f acc[4];
#pragma unroll
        for (int t = 0; t < 4; ++t) { acc[t] = {}; pf_tile(Wt0 + t * 16 * DD, lane); }
#pragma unroll
        for (int kk = 0; kk < 8; ++kk) {
            bv16 a = ld_a(Kgbf, 0, kk * 32, lane);
#pragma unroll
            for (int t = 0; t < 4; ++t) {
                bv16 b = ld_b(Wt0 + t * 16 * DD, kk * 32, lane);
                acc[t] = wmma_bf16(a, b, acc[t]);
            }
        }
#pragma unroll
        for (int t = 0; t < 4; ++t) {
            int n = (wave * 4 + t) * 16 + nlo;
#pragma unroll
            for (int i = 0; i < 8; ++i) KVpbf[(mh + i) * 512 + n] = (bf16_t)acc[t][i];
        }
    }
    __syncthreads();

    // ---- S7: attention. G[lb][h][j][k] = (SLq_h[j] . Kp_h[k]) / 16; att = softmax_k(S2 @ G) ----
    if (tid < 144) {
        int lb = tid / 36, rem = tid % 36, h = rem / 9, jk = rem % 9;
        int j = jk / 3, k = jk - j * 3;
        const bf16_t* qp = SLqbf + (lb * 3 + j) * DD + h * 64;
        const bf16_t* kp = KVpbf + (lb * 3 + k) * 512 + h * 64;
        float s = 0.f;
#pragma unroll 8
        for (int d = 0; d < 64; ++d) s += (float)qp[d] * (float)kp[d];
        Gm[tid] = s * 0.0625f;   // 1/sqrt(256)
    }
    __syncthreads();
    for (int t2 = tid; t2 < 400; t2 += 256) {
        int lb = t2 / 100, rem = t2 % 100, h = rem / 25, qq = rem - h * 25;
        const float* g = Gm + lb * 36 + h * 9;
        float s20 = S2[qq * 3], s21 = S2[qq * 3 + 1], s22 = S2[qq * 3 + 2];
        float sc0 = s20 * g[0] + s21 * g[3] + s22 * g[6];
        float sc1 = s20 * g[1] + s21 * g[4] + s22 * g[7];
        float sc2 = s20 * g[2] + s21 * g[5] + s22 * g[8];
        float mx = fmaxf(sc0, fmaxf(sc1, sc2));
        float e0 = __expf(sc0 - mx), e1 = __expf(sc1 - mx), e2 = __expf(sc2 - mx);
        float rs = 1.f / (e0 + e1 + e2);
        float* ap = att + ((lb * 4 + h) * 25 + qq) * 3;
        ap[0] = e0 * rs; ap[1] = e1 * rs; ap[2] = e2 * rs;
    }
    __syncthreads();

    // ---- S8: O2[q] = S2@SLq + att@Vp, LN(g21,b21) -> O2bf ----
    for (int r = wave; r < O2P; r += 8) {
        int cb = lane * 8;
        if (r < O2R) {
            int lb = r / 25, qq = r - lb * 25;
            float s20 = S2[qq * 3], s21 = S2[qq * 3 + 1], s22 = S2[qq * 3 + 2];
            const bf16_t* sl0 = SLqbf + (lb * 3 + 0) * DD;
            const bf16_t* sl1 = SLqbf + (lb * 3 + 1) * DD;
            const bf16_t* sl2 = SLqbf + (lb * 3 + 2) * DD;
            const bf16_t* v0 = KVpbf + (lb * 3 + 0) * 512 + 256;
            const bf16_t* v1 = KVpbf + (lb * 3 + 1) * 512 + 256;
            const bf16_t* v2 = KVpbf + (lb * 3 + 2) * 512 + 256;
            float vals[8]; float s = 0.f, q = 0.f;
#pragma unroll
            for (int j = 0; j < 8; ++j) {
                int c = cb + j;
                int h = c >> 6;
                const float* ap = att + ((lb * 4 + h) * 25 + qq) * 3;
                float o = s20 * (float)sl0[c] + s21 * (float)sl1[c] + s22 * (float)sl2[c]
                        + ap[0] * (float)v0[c] + ap[1] * (float)v1[c] + ap[2] * (float)v2[c];
                vals[j] = o; s += o; q += o * o;
            }
            wave_reduce2(s, q);
            float mean = s * (1.f / 256.f);
            float rinv = rsqrtf(q * (1.f / 256.f) - mean * mean + 1e-5f);
#pragma unroll
            for (int j = 0; j < 8; ++j) {
                int c = cb + j;
                O2bf[r * DD + c] = (bf16_t)((vals[j] - mean) * rinv * g21[c] + b21[c]);
            }
        } else {
#pragma unroll
            for (int j = 0; j < 8; ++j) O2bf[r * DD + cb + j] = (bf16_t)0.f;
        }
    }
    __syncthreads();

    // ---- S9 (G4): T2 = O2bf @ Wo2, out = LN(O2 + relu(T2), g22,b22) ----
    // All 7 M-tile accumulators live in registers: Wo2^T streams from L2 exactly once per WG.
    {
        const bf16_t* Wt0 = wts + OFF_WO2T + (wave * 2 + 0) * 16 * DD;
        const bf16_t* Wt1 = wts + OFF_WO2T + (wave * 2 + 1) * 16 * DD;
        pf_tile(Wt0, lane);
        pf_tile(Wt1, lane);
        v8f acc[2][7];
#pragma unroll
        for (int t = 0; t < 2; ++t)
#pragma unroll
            for (int mt = 0; mt < 7; ++mt) acc[t][mt] = {};

#pragma unroll
        for (int kk = 0; kk < 8; ++kk) {
            bv16 b0 = ld_b(Wt0, kk * 32, lane);
            bv16 b1 = ld_b(Wt1, kk * 32, lane);
#pragma unroll
            for (int mt = 0; mt < 7; ++mt) {
                bv16 a = ld_a(O2bf, mt * 16, kk * 32, lane);
                acc[0][mt] = wmma_bf16(a, b0, acc[0][mt]);
                acc[1][mt] = wmma_bf16(a, b1, acc[1][mt]);
            }
        }

#pragma unroll
        for (int chunk = 0; chunk < 4; ++chunk) {
            const int mtbase = chunk * 2;
            const int nmt = (chunk < 3) ? 2 : 1;
#pragma unroll
            for (int t = 0; t < 2; ++t) {
                int n = (wave * 2 + t) * 16 + nlo;
#pragma unroll
                for (int mm = 0; mm < 2; ++mm) {
                    if (mm < nmt) {
#pragma unroll
                        for (int i = 0; i < 8; ++i)
                            stage[(mm * 16 + mh + i) * SS4 + n] = acc[t][mtbase + mm][i];
                    }
                }
            }
            __syncthreads();
            for (int lr = wave; lr < nmt * 16; lr += 8) {
                int r = mtbase * 16 + lr;
                if (r < O2R) {
                    int cb = lane * 8;
                    float vals[8]; float s = 0.f, q = 0.f;
#pragma unroll
                    for (int j = 0; j < 8; ++j) {
                        int c = cb + j;
                        float o = (float)O2bf[r * DD + c] + fmaxf(stage[lr * SS4 + c], 0.f);
                        vals[j] = o; s += o; q += o * o;
                    }
                    wave_reduce2(s, q);
                    float mean = s * (1.f / 256.f);
                    float rinv = rsqrtf(q * (1.f / 256.f) - mean * mean + 1e-5f);
                    size_t base = ((size_t)blockIdx.x * O2R + r) * DD;
#pragma unroll
                    for (int j = 0; j < 8; ++j) {
                        int c = cb + j;
                        out[base + c] = (vals[j] - mean) * rinv * g22[c] + b22[c];
                    }
                }
            }
            __syncthreads();
        }
    }
}

extern "C" void kernel_launch(void* const* d_in, const int* in_sizes, int n_in,
                              void* d_out, int out_size, void* d_ws, size_t ws_size,
                              hipStream_t stream) {
    (void)n_in; (void)out_size; (void)ws_size;
    const float* X    = (const float*)d_in[0];
    const float* S1   = (const float*)d_in[1];
    const float* S2   = (const float*)d_in[2];
    const float* adj2 = (const float*)d_in[4];
    const float* Wq1  = (const float*)d_in[5];
    const float* Wv1  = (const float*)d_in[7];   // Wk1 (d_in[6]) is dead: softmax over 1 key == 1
    const float* Wo1  = (const float*)d_in[8];
    const float* g11  = (const float*)d_in[9];
    const float* b11  = (const float*)d_in[10];
    const float* g12  = (const float*)d_in[11];
    const float* b12  = (const float*)d_in[12];
    const float* Wq2  = (const float*)d_in[13];
    const float* Wk2  = (const float*)d_in[14];
    const float* Wv2  = (const float*)d_in[15];
    const float* Wo2  = (const float*)d_in[16];
    const float* g21  = (const float*)d_in[17];
    const float* b21  = (const float*)d_in[18];
    const float* g22  = (const float*)d_in[19];
    const float* b22  = (const float*)d_in[20];
    float*  out = (float*)d_out;
    bf16_t* ws  = (bf16_t*)d_ws;

    int Bn = in_sizes[0] / DD;   // X is [B,1,256]

    // 7 matrices x 64 tiles, transposed+converted to bf16 with coalesced reads/writes
    prep_weights<<<7 * 64, 256, 0, stream>>>(Wq1, Wv1, Wo1, Wq2, Wk2, Wv2, Wo2, ws);

    (void)hipFuncSetAttribute((const void*)fused_gtd_kernel,
                              hipFuncAttributeMaxDynamicSharedMemorySize, SMEM_BYTES);
    fused_gtd_kernel<<<Bn / NB, 256, SMEM_BYTES, stream>>>(
        X, S1, S2, adj2, g11, b11, g12, b12, g21, b21, g22, b22, ws, out);
}
